// MoProGCN_65867618451817
// MI455X (gfx1250) — matrix operand order repaired and checked by reading, not compiled, not run on hardware
//
#include <hip/hip_runtime.h>

// ---------------------------------------------------------------------------
// MoPro-GCN forward for MI455X (gfx1250, wave32, WMMA).
// GEMMs: bf16 v_wmma_f32_16x16x32_bf16, fp32 accumulation, LDS-tiled with
// double-buffered GLOBAL_LOAD_ASYNC_TO_LDS_B128 staging (ASYNCcnt-tracked).
// ---------------------------------------------------------------------------

#define B_ 256
#define N_ 5
#define D_ 2048
#define H_ 1024
#define C_ 1000

// GEMM block tile
#define BM 128
#define BN 64
#define BK 32

typedef __attribute__((ext_vector_type(16))) __bf16          v16bf;
typedef __attribute__((ext_vector_type(8)))  float           v8f;
typedef __attribute__((ext_vector_type(8)))  unsigned short  u16x8;

union FragU {            // 32-byte bf16 fragment, read as 2x b128
    v16bf h;
    u16x8 v[2];
};

static __device__ __forceinline__ unsigned short f32_to_bf16(float f) {
    union { float f; unsigned u; } un; un.f = f;
    unsigned r = un.u + 0x7FFFu + ((un.u >> 16) & 1u);   // round-to-nearest-even
    return (unsigned short)(r >> 16);
}

// async copy of 16 bytes: global -> LDS (ASYNCcnt-tracked, per-lane addresses)
static __device__ __forceinline__ void async_g2l_b128(const void* gsrc, void* ldst) {
    unsigned       lds = (unsigned)(size_t)ldst;          // low 32 bits = LDS offset
    unsigned long long g = (unsigned long long)(size_t)gsrc;
    asm volatile("global_load_async_to_lds_b128 %0, %1, off"
                 :: "v"(lds), "v"(g) : "memory");
}

static __device__ __forceinline__ void wait_async0() {
#if __has_builtin(__builtin_amdgcn_s_wait_asynccnt)
    __builtin_amdgcn_s_wait_asynccnt(0);
#else
    asm volatile("s_wait_asynccnt 0x0" ::: "memory");
#endif
}

// ---------------------------------------------------------------------------
// Transpose + fp32 -> bf16:  src [R, C] fp32  ->  dst [C, R] bf16
// ---------------------------------------------------------------------------
__global__ void transpose_to_bf16_kernel(const float* __restrict__ src,
                                         unsigned short* __restrict__ dst,
                                         int R, int Cc) {
    int idx = blockIdx.x * blockDim.x + threadIdx.x;
    if (idx >= R * Cc) return;
    int r = idx / Cc;
    int c = idx - r * Cc;
    dst[(size_t)c * R + r] = f32_to_bf16(src[idx]);
}

// ---------------------------------------------------------------------------
// Adjacency mix (5x5) + convert:  dst[b,n,d] = bf16( sum_m adj[n,m]*src[b,m,d] )
// ---------------------------------------------------------------------------
__global__ void mix_adj_bf16_kernel(const float* __restrict__ src,
                                    const float* __restrict__ adj,
                                    unsigned short* __restrict__ dst,
                                    int Dim, int total) {
    int idx = blockIdx.x * blockDim.x + threadIdx.x;
    if (idx >= total) return;
    int d = idx % Dim;
    int n = (idx / Dim) % N_;
    int b = idx / (Dim * N_);
    float s = 0.f;
#pragma unroll
    for (int m = 0; m < N_; ++m)
        s += adj[n * N_ + m] * src[((size_t)b * N_ + m) * Dim + d];
    dst[idx] = f32_to_bf16(s);
}

// ---------------------------------------------------------------------------
// LDS-tiled WMMA GEMM:  C[M,N] = act( A[M,K](bf16) @ BT[N,K]^T(bf16) + bias )
//
// Block = 256 threads = 8 waves in a 4(M) x 2(N) grid; each wave owns a 32x32
// patch = four 16x16 fp32 accumulators.  Per K-step of 32:
//   - all 256 threads issue 3x global_load_async_to_lds_b128 (A 8KB + B 4KB)
//     into the non-active LDS buffer (double-buffered, 24KB total),
//   - s_wait_asynccnt 0 + barrier, then waves read fragments with ds_load_b128
//     following the CDNA5 16-bit A layout (ISA 7.12.2):
//       lanes 0-15 : row=l15,  K = k+0..7  and k+16..23
//       lanes 16-31: row=l15,  K = k+8..15 and k+24..31
// Requires M % 128 == 0 (true for all call sites); N may be ragged (clamped).
// ---------------------------------------------------------------------------
__global__ void wmma_gemm_bf16_kernel(const unsigned short* __restrict__ A,
                                      const unsigned short* __restrict__ BT,
                                      float* __restrict__ Cf,           // fp32 out (or null)
                                      unsigned short* __restrict__ Cb,  // bf16 out (or null)
                                      const float* __restrict__ bias,   // [N] or null
                                      int M, int N, int K, int relu) {
    __shared__ unsigned short As[2][BM * BK];   // 2 x 8KB
    __shared__ unsigned short Bs[2][BN * BK];   // 2 x 4KB

    const int tilesN = (N + BN - 1) / BN;
    const int bm = blockIdx.x / tilesN;
    const int bn = blockIdx.x - bm * tilesN;
    const int m0 = bm * BM;
    const int n0 = bn * BN;

    const int t    = threadIdx.x;
    const int wave = t >> 5;
    const int wm   = wave & 3;        // 0..3 -> M offset wm*32
    const int wn   = wave >> 2;       // 0..1 -> N offset wn*32
    const int lane = t & 31;
    const int half = lane >> 4;
    const int l15  = lane & 15;

    const int nk = K / BK;

    // --- async stage of one K-slab into LDS buffer `buf` ---
    auto issue = [&](int buf, int kt) {
        const int k0 = kt * BK;
#pragma unroll
        for (int i = 0; i < 2; ++i) {                 // A: 128 rows x 32 cols
            int c   = t + i * 256;                    // chunk of 8 bf16
            int row = c >> 2;
            int cc  = (c & 3) * 8;
            async_g2l_b128(A + (size_t)(m0 + row) * K + k0 + cc,
                           &As[buf][row * BK + cc]);
        }
        {                                             // B: 64 rows x 32 cols
            int row = t >> 2;
            int cc  = (t & 3) * 8;
            int gr  = n0 + row; if (gr >= N) gr = N - 1;
            async_g2l_b128(BT + (size_t)gr * K + k0 + cc,
                           &Bs[buf][row * BK + cc]);
        }
    };

    v8f acc00 = {}, acc01 = {}, acc10 = {}, acc11 = {};

    issue(0, 0);
    int buf = 0;
    for (int kt = 0; kt < nk; ++kt) {
        wait_async0();                 // own async copies done
        __syncthreads();               // everyone's copies visible
        if (kt + 1 < nk) issue(buf ^ 1, kt + 1);

        FragU a0, a1, b0, b1;
        const unsigned short* ap = &As[buf][(wm * 32 + l15) * BK + half * 8];
        a0.v[0] = *(const u16x8*)(ap);
        a0.v[1] = *(const u16x8*)(ap + 16);
        a1.v[0] = *(const u16x8*)(ap + 16 * BK);
        a1.v[1] = *(const u16x8*)(ap + 16 * BK + 16);
        const unsigned short* bp = &Bs[buf][(wn * 32 + l15) * BK + half * 8];
        b0.v[0] = *(const u16x8*)(bp);
        b0.v[1] = *(const u16x8*)(bp + 16);
        b1.v[0] = *(const u16x8*)(bp + 16 * BK);
        b1.v[1] = *(const u16x8*)(bp + 16 * BK + 16);

        acc00 = __builtin_amdgcn_wmma_f32_16x16x32_bf16(false, a0.h, false, b0.h,
                                                        (short)0, acc00, false, false);
        acc01 = __builtin_amdgcn_wmma_f32_16x16x32_bf16(false, a0.h, false, b1.h,
                                                        (short)0, acc01, false, false);
        acc10 = __builtin_amdgcn_wmma_f32_16x16x32_bf16(false, a1.h, false, b0.h,
                                                        (short)0, acc10, false, false);
        acc11 = __builtin_amdgcn_wmma_f32_16x16x32_bf16(false, a1.h, false, b1.h,
                                                        (short)0, acc11, false, false);

        __syncthreads();               // all waves done reading buf
        buf ^= 1;
    }

    // --- epilogue: C layout is VGPR r -> row (r + 8*half), col = l15 ---
    v8f accs[2][2] = { { acc00, acc01 }, { acc10, acc11 } };
#pragma unroll
    for (int mi = 0; mi < 2; ++mi) {
#pragma unroll
        for (int ni = 0; ni < 2; ++ni) {
            const int col = n0 + wn * 32 + ni * 16 + l15;
            if (col >= N) continue;
            const float bv = (bias != nullptr) ? bias[col] : 0.f;
#pragma unroll
            for (int r = 0; r < 8; ++r) {
                const int row = m0 + wm * 32 + mi * 16 + half * 8 + r;
                float v = accs[mi][ni][r] + bv;
                if (relu && v < 0.f) v = 0.f;
                size_t o = (size_t)row * N + col;
                if (Cf) Cf[o] = v;
                if (Cb) Cb[o] = f32_to_bf16(v);
            }
        }
    }
}

// ---------------------------------------------------------------------------
// Sequential-equivalent prototype EMA update + L2 normalize over node axis.
// One block per class c; batch scan is block-uniform (targets staged in LDS).
// ---------------------------------------------------------------------------
__global__ void proto_update_kernel(const float* __restrict__ protos,
                                    const float* __restrict__ x,
                                    const int* __restrict__ target,
                                    float* __restrict__ out) {
    __shared__ int tgt[B_];
    const int c = blockIdx.x;
    tgt[threadIdx.x] = target[threadIdx.x];
    __syncthreads();

    for (int j = 0; j < D_ / 256; ++j) {
        const int d = threadIdx.x + j * 256;
        float p[N_];
#pragma unroll
        for (int n = 0; n < N_; ++n)
            p[n] = protos[((size_t)c * N_ + n) * D_ + d];

        for (int b = 0; b < B_; ++b) {
            if (tgt[b] == c) {                    // uniform per block -> scalar branch
#pragma unroll
                for (int n = 0; n < N_; ++n)
                    p[n] = p[n] * 0.999f + 0.001f * x[((size_t)b * N_ + n) * D_ + d];
            }
        }

        float s = 0.f;
#pragma unroll
        for (int n = 0; n < N_; ++n) s += p[n] * p[n];
        const float inv = 1.0f / fmaxf(sqrtf(s), 1e-12f);
#pragma unroll
        for (int n = 0; n < N_; ++n)
            out[((size_t)c * N_ + n) * D_ + d] = p[n] * inv;
    }
}

// ---------------------------------------------------------------------------
// Host launcher
// ---------------------------------------------------------------------------
static inline size_t align_up256(size_t v) { return (v + 255) & ~(size_t)255; }

extern "C" void kernel_launch(void* const* d_in, const int* in_sizes, int n_in,
                              void* d_out, int out_size, void* d_ws, size_t ws_size,
                              hipStream_t stream) {
    (void)in_sizes; (void)n_in; (void)out_size; (void)ws_size;

    const float* x      = (const float*)d_in[0];   // [B,N,D]
    const int*   target = (const int*)  d_in[1];   // [B]
    const float* protos = (const float*)d_in[2];   // [C,N,D]
    const float* adj    = (const float*)d_in[3];   // [N,N]
    const float* W1     = (const float*)d_in[4];   // [D,H]
    const float* W2     = (const float*)d_in[5];   // [H,D]
    const float* Wg     = (const float*)d_in[6];   // [N*D,D]
    const float* bg     = (const float*)d_in[7];   // [D]
    const float* Wc     = (const float*)d_in[8];   // [D,C]
    const float* bc     = (const float*)d_in[9];   // [C]

    float* pred     = (float*)d_out;               // [B,C]
    float* outProto = pred + (size_t)B_ * C_;      // [C,N,D]

    // ---- workspace carve-out (bf16 stored as u16) ----
    char* ws = (char*)d_ws;
    size_t off = 0;
    auto carve = [&](size_t bytes) -> void* {
        void* p = ws + off; off = align_up256(off + bytes); return p;
    };
    unsigned short* axB    = (unsigned short*)carve((size_t)B_ * N_ * D_ * 2); // [1280,2048]
    unsigned short* W1T    = (unsigned short*)carve((size_t)H_ * D_ * 2);      // [1024,2048]
    float*          hF     = (float*)         carve((size_t)B_ * N_ * H_ * 4); // [1280,1024]
    unsigned short* ahB    = (unsigned short*)carve((size_t)B_ * N_ * H_ * 2); // [1280,1024]
    unsigned short* W2T    = (unsigned short*)carve((size_t)D_ * H_ * 2);      // [2048,1024]
    unsigned short* nodesB = (unsigned short*)carve((size_t)B_ * N_ * D_ * 2); // [256,10240]
    unsigned short* WgT    = (unsigned short*)carve((size_t)D_ * N_ * D_ * 2); // [2048,10240]
    unsigned short* gB     = (unsigned short*)carve((size_t)B_ * D_ * 2);      // [256,2048]
    unsigned short* WcT    = (unsigned short*)carve((size_t)C_ * D_ * 2);      // [1000,2048]

    const int T = 256;
    auto blk  = [](long long n, int t) { return (int)((n + t - 1) / t); };
    auto gemm = [&](const unsigned short* Ain, const unsigned short* Bin,
                    float* Cf, unsigned short* Cb, const float* bias,
                    int M, int N, int K, int relu) {
        int grid = (M / BM) * ((N + BN - 1) / BN);
        wmma_gemm_bf16_kernel<<<grid, T, 0, stream>>>(Ain, Bin, Cf, Cb, bias, M, N, K, relu);
    };

    // ---- weight transposes to bf16 [N,K] ----
    transpose_to_bf16_kernel<<<blk((long long)D_ * H_, T), T, 0, stream>>>(W1, W1T, D_, H_);
    transpose_to_bf16_kernel<<<blk((long long)H_ * D_, T), T, 0, stream>>>(W2, W2T, H_, D_);
    transpose_to_bf16_kernel<<<blk((long long)N_ * D_ * D_, T), T, 0, stream>>>(Wg, WgT, N_ * D_, D_);
    transpose_to_bf16_kernel<<<blk((long long)D_ * C_, T), T, 0, stream>>>(Wc, WcT, D_, C_);

    // ---- ax = adj @ x  (bf16) ----
    mix_adj_bf16_kernel<<<blk((long long)B_ * N_ * D_, T), T, 0, stream>>>(
        x, adj, axB, D_, B_ * N_ * D_);

    // ---- h = relu(ax @ W1)  : [1280,1024] fp32 ----
    gemm(axB, W1T, hF, nullptr, nullptr, B_ * N_, H_, D_, 1);

    // ---- ah = adj @ h  (bf16) ----
    mix_adj_bf16_kernel<<<blk((long long)B_ * N_ * H_, T), T, 0, stream>>>(
        hF, adj, ahB, H_, B_ * N_ * H_);

    // ---- nodes = ah @ W2 : [1280,2048] bf16 (== [256,10240] reshaped) ----
    gemm(ahB, W2T, nullptr, nodesB, nullptr, B_ * N_, D_, H_, 0);

    // ---- g = nodes @ Wg + bg : [256,2048] bf16 ----
    gemm(nodesB, WgT, nullptr, gB, bg, B_, D_, N_ * D_, 0);

    // ---- pred = g @ Wc + bc : [256,1000] fp32 -> d_out ----
    gemm(gB, WcT, pred, nullptr, bc, B_, C_, D_, 0);

    // ---- prototype EMA scatter + normalize -> d_out tail ----
    proto_update_kernel<<<C_, 256, 0, stream>>>(protos, x, target, outProto);
}
